// MessagePassingLayer_69621419868955
// MI455X (gfx1250) — compile-verified, matrix operand order checked by reading, not run on hardware
//
#include <hip/hip_runtime.h>
#include <stdint.h>

#define LN_EPS 1e-5f

typedef __attribute__((ext_vector_type(16))) __bf16 bf16x16;
typedef __attribute__((ext_vector_type(8)))  float  floatx8;

union Frag {
    uint32_t u[8];
    uint4    q[2];
    bf16x16  v;
};

// round-to-nearest-even float -> bf16, packed pair into a dword
__device__ __forceinline__ uint32_t pack_bf2(float lo, float hi) {
    uint32_t ul = __float_as_uint(lo);
    uint32_t uh = __float_as_uint(hi);
    uint32_t bl = (ul + 0x7FFFu + ((ul >> 16) & 1u)) >> 16;
    uint32_t bh = (uh + 0x7FFFu + ((uh >> 16) & 1u)) >> 16;
    return (bl & 0xFFFFu) | (bh << 16);
}

// A-fragment (16x32 bf16, M x K) from LDS row-major bf16 (strideDw dwords per row).
// ISA layout: lanes 0-15 -> M=lane, K={0..7,16..23}; lanes 16-31 -> K={8..15,24..31}.
__device__ __forceinline__ void load_a_frag(const uint32_t* sA, int strideDw, int mt, int kt,
                                            int lane, Frag& f) {
    const uint32_t* p = sA + (mt * 16 + (lane & 15)) * strideDw + kt * 16 + ((lane & 16) ? 4 : 0);
    f.q[0] = *(const uint4*)(p);
    f.q[1] = *(const uint4*)(p + 8);
}

// B-fragment from pre-packed weights: layout [kt][nt][lane][8 dwords] (lane-contiguous 32B)
__device__ __forceinline__ void load_b_frag(const uint32_t* __restrict__ wp, int NT, int kt,
                                            int nt, int lane, Frag& f) {
    const uint4* p = (const uint4*)(wp + (((kt * NT + nt) * 32 + lane) << 3));
    f.q[0] = p[0];
    f.q[1] = p[1];
}

__global__ __launch_bounds__(256) void zero_f32_kernel(float* __restrict__ p, long long n) {
    long long i = (long long)blockIdx.x * 256 + threadIdx.x;
    if (i < n) p[i] = 0.0f;
}

// Repack fp32 W[K,N] (row-major) into bf16 WMMA B-fragment order.
__global__ __launch_bounds__(256) void pack_weights_kernel(const float* __restrict__ W,
                                                           uint32_t* __restrict__ out,
                                                           int K, int N) {
    int NT = N >> 4;
    int total = (K >> 5) * NT * 256;
    for (int idx = blockIdx.x * 256 + threadIdx.x; idx < total; idx += gridDim.x * 256) {
        int v    = idx & 7;
        int lane = (idx >> 3) & 31;
        int tile = idx >> 8;
        int nt = tile % NT;
        int kt = tile / NT;
        int n  = nt * 16 + (lane & 15);
        int khalf = (lane & 16) ? 8 : 0;
        int kk = (v < 4) ? (khalf + 2 * v) : (16 + khalf + 2 * (v - 4));
        int k  = kt * 32 + kk;
        out[idx] = pack_bf2(W[(long long)k * N + n], W[(long long)(k + 1) * N + n]);
    }
}

// GEMM1: [32 x K] (bf16 LDS) x packed W[K x 256] -> sH [32 x 256] fp32.
// Wave w owns nt {2w, 2w+1} x mt {0,1}: each B fragment loaded once per block, reused 2x.
__device__ __forceinline__ void gemm1_32x256(const uint32_t* __restrict__ sX, int strideDw, int KT,
                                             const uint32_t* __restrict__ pW1,
                                             float* __restrict__ sH, int tid) {
    const int lane = tid & 31;
    const int ntb  = (tid >> 5) * 2;
    floatx8 acc[2][2] = {};
    for (int kt = 0; kt < KT; ++kt) {
        Frag a0, a1;
        load_a_frag(sX, strideDw, 0, kt, lane, a0);
        load_a_frag(sX, strideDw, 1, kt, lane, a1);
        #pragma unroll
        for (int j = 0; j < 2; ++j) {
            Frag b; load_b_frag(pW1, 16, kt, ntb + j, lane, b);
            acc[0][j] = __builtin_amdgcn_wmma_f32_16x16x32_bf16(
                false, a0.v, false, b.v, (short)0, acc[0][j], false, false);
            acc[1][j] = __builtin_amdgcn_wmma_f32_16x16x32_bf16(
                false, a1.v, false, b.v, (short)0, acc[1][j], false, false);
        }
    }
    #pragma unroll
    for (int m = 0; m < 2; ++m) {
        #pragma unroll
        for (int j = 0; j < 2; ++j) {
            int col = (ntb + j) * 16 + (lane & 15);
            #pragma unroll
            for (int r = 0; r < 8; ++r) {
                int row = m * 16 + r + ((lane & 16) >> 1);   // +8 for upper half-wave
                sH[row * 256 + col] = acc[m][j][r];
            }
        }
    }
}

// bias add -> LayerNorm -> SiLU -> bf16 repack into sHb [32 x 128 dwords].
// 8 lanes per row, shfl_xor reduction (wave32, groups of 8 stay in-wave).
__device__ __forceinline__ void ln_silu_pack(const float* __restrict__ sH,
                                             uint32_t* __restrict__ sHb,
                                             const float* __restrict__ b1,
                                             const float* __restrict__ g1,
                                             const float* __restrict__ be1,
                                             int tid) {
    const int row = tid >> 3;
    const int sub = tid & 7;
    const float* hr = sH + row * 256 + sub * 32;
    const float* bb = b1 + sub * 32;
    float vals[32];
    float s = 0.f, ss = 0.f;
    #pragma unroll
    for (int c = 0; c < 32; ++c) {
        float x = hr[c] + bb[c];
        vals[c] = x; s += x; ss += x * x;
    }
    s += __shfl_xor(s, 1);  ss += __shfl_xor(ss, 1);
    s += __shfl_xor(s, 2);  ss += __shfl_xor(ss, 2);
    s += __shfl_xor(s, 4);  ss += __shfl_xor(ss, 4);
    float mean = s * (1.0f / 256.0f);
    float var  = ss * (1.0f / 256.0f) - mean * mean;
    float rstd = rsqrtf(var + LN_EPS);
    #pragma unroll
    for (int c = 0; c < 32; c += 2) {
        int col = sub * 32 + c;
        float x0 = (vals[c]     - mean) * rstd * g1[col]     + be1[col];
        float x1 = (vals[c + 1] - mean) * rstd * g1[col + 1] + be1[col + 1];
        x0 = x0 / (1.0f + __expf(-x0));
        x1 = x1 / (1.0f + __expf(-x1));
        sHb[row * 128 + (col >> 1)] = pack_bf2(x0, x1);
    }
}

// Fused edge path: gather -> GEMM1 -> LN/SiLU -> GEMM2 -> +bias +residual -> store + scatter-add
__global__ __launch_bounds__(256) void edge_mlp_kernel(
    const float* __restrict__ nodeF, const float* __restrict__ edgeF,
    const long long* __restrict__ senders, const long long* __restrict__ receivers,
    const uint32_t* __restrict__ pW1, const float* __restrict__ eb1,
    const float* __restrict__ eg1, const float* __restrict__ ebeta1,
    const uint32_t* __restrict__ pW2, const float* __restrict__ eb2,
    float* __restrict__ edgeOut, float* __restrict__ agg, int E)
{
    __shared__ uint32_t sX[32 * 192];   // 24KB bf16 input 32x384 (reused as sHb after GEMM1)
    __shared__ float    sH[32 * 256];   // 32KB fp32 hidden
    __shared__ int      sSend[32];
    __shared__ int      sRecv[32];
    uint32_t* sHb = sX;                 // 16KB alias, safe: last sX read is in GEMM1

    const int tid  = threadIdx.x;
    const int lane = tid & 31;
    const long long e0 = (long long)blockIdx.x * 32;

    if (tid == 0) __builtin_prefetch(pW1, 0, 1);

    if (tid < 32)      sSend[tid]      = (e0 + tid      < E) ? (int)senders[e0 + tid]          : 0;
    else if (tid < 64) sRecv[tid - 32] = (e0 + tid - 32 < E) ? (int)receivers[e0 + (tid - 32)] : 0;
    __syncthreads();

    // stage gathered+concatenated input as bf16
    for (int i = tid; i < 32 * 192; i += 256) {
        int row = i / 192;
        int p   = i - row * 192;
        int c0  = p * 2;
        long long e = e0 + row;
        float2 v = {0.f, 0.f};
        if (e < E) {
            const float* src;
            if (c0 < 128)      src = nodeF + (long long)sSend[row] * 128 + c0;
            else if (c0 < 256) src = nodeF + (long long)sRecv[row] * 128 + (c0 - 128);
            else               src = edgeF + e * 128 + (c0 - 256);
            v = *(const float2*)src;
        }
        sX[i] = pack_bf2(v.x, v.y);
    }
    __syncthreads();

    gemm1_32x256(sX, 192, 12, pW1, sH, tid);    // K = 384
    __syncthreads();

    ln_silu_pack(sH, sHb, eb1, eg1, ebeta1, tid);
    __syncthreads();

    // GEMM2: [32x256] x [256x128], wave w -> nt = w, mt {0,1}; B loaded once, used 2x
    {
        const int w = tid >> 5;
        floatx8 acc[2] = {};
        for (int kt = 0; kt < 8; ++kt) {
            Frag a0, a1, b;
            load_a_frag(sHb, 128, 0, kt, lane, a0);
            load_a_frag(sHb, 128, 1, kt, lane, a1);
            load_b_frag(pW2, 8, kt, w, lane, b);
            acc[0] = __builtin_amdgcn_wmma_f32_16x16x32_bf16(
                false, a0.v, false, b.v, (short)0, acc[0], false, false);
            acc[1] = __builtin_amdgcn_wmma_f32_16x16x32_bf16(
                false, a1.v, false, b.v, (short)0, acc[1], false, false);
        }
        int col = w * 16 + (lane & 15);
        float bias = eb2[col];
        #pragma unroll
        for (int m = 0; m < 2; ++m) {
            #pragma unroll
            for (int r = 0; r < 8; ++r) {
                int row = m * 16 + r + ((lane & 16) >> 1);
                long long e = e0 + row;
                if (e < E) {
                    float v = acc[m][r] + bias + edgeF[e * 128 + col];
                    edgeOut[e * 128 + col] = v;
                    atomicAdd(agg + (long long)sRecv[row] * 128 + col, v);
                }
            }
        }
    }
}

// Fused node path: [node||agg] -> GEMM1 -> LN/SiLU -> GEMM2 -> +bias +residual -> store
__global__ __launch_bounds__(256) void node_mlp_kernel(
    const float* __restrict__ nodeF, const float* __restrict__ agg,
    const uint32_t* __restrict__ pW1, const float* __restrict__ nb1,
    const float* __restrict__ ng1, const float* __restrict__ nbeta1,
    const uint32_t* __restrict__ pW2, const float* __restrict__ nb2,
    float* __restrict__ nodeOut, int N)
{
    __shared__ uint32_t sX[32 * 128];   // 16KB bf16 input 32x256 (reused as sHb)
    __shared__ float    sH[32 * 256];   // 32KB
    uint32_t* sHb = sX;

    const int tid  = threadIdx.x;
    const int lane = tid & 31;
    const long long n0 = (long long)blockIdx.x * 32;

    if (tid == 0) __builtin_prefetch(pW1, 0, 1);

    for (int i = tid; i < 32 * 128; i += 256) {
        int row = i >> 7;
        int p   = i & 127;
        int c0  = p * 2;
        long long n = n0 + row;
        float2 v = {0.f, 0.f};
        if (n < N) {
            const float* src = (c0 < 128) ? (nodeF + n * 128 + c0)
                                          : (agg   + n * 128 + (c0 - 128));
            v = *(const float2*)src;
        }
        sX[i] = pack_bf2(v.x, v.y);
    }
    __syncthreads();

    gemm1_32x256(sX, 128, 8, pW1, sH, tid);     // K = 256
    __syncthreads();

    ln_silu_pack(sH, sHb, nb1, ng1, nbeta1, tid);
    __syncthreads();

    {
        const int w = tid >> 5;
        floatx8 acc[2] = {};
        for (int kt = 0; kt < 8; ++kt) {
            Frag a0, a1, b;
            load_a_frag(sHb, 128, 0, kt, lane, a0);
            load_a_frag(sHb, 128, 1, kt, lane, a1);
            load_b_frag(pW2, 8, kt, w, lane, b);
            acc[0] = __builtin_amdgcn_wmma_f32_16x16x32_bf16(
                false, a0.v, false, b.v, (short)0, acc[0], false, false);
            acc[1] = __builtin_amdgcn_wmma_f32_16x16x32_bf16(
                false, a1.v, false, b.v, (short)0, acc[1], false, false);
        }
        int col = w * 16 + (lane & 15);
        float bias = nb2[col];
        #pragma unroll
        for (int m = 0; m < 2; ++m) {
            #pragma unroll
            for (int r = 0; r < 8; ++r) {
                int row = m * 16 + r + ((lane & 16) >> 1);
                long long n = n0 + row;
                if (n < N) {
                    float v = acc[m][r] + bias + nodeF[n * 128 + col];
                    nodeOut[n * 128 + col] = v;
                }
            }
        }
    }
}

extern "C" void kernel_launch(void* const* d_in, const int* in_sizes, int n_in,
                              void* d_out, int out_size, void* d_ws, size_t ws_size,
                              hipStream_t stream) {
    const int N = in_sizes[0] / 128;   // 50000
    const int E = in_sizes[1] / 128;   // 600000

    const float*     nodeF  = (const float*)d_in[0];
    const float*     edgeF  = (const float*)d_in[1];
    const long long* eidx   = (const long long*)d_in[2];
    const float*     eW1    = (const float*)d_in[3];
    const float*     eb1    = (const float*)d_in[4];
    const float*     eg1    = (const float*)d_in[5];
    const float*     ebeta1 = (const float*)d_in[6];
    const float*     eW2    = (const float*)d_in[7];
    const float*     eb2    = (const float*)d_in[8];
    const float*     nW1    = (const float*)d_in[9];
    const float*     nb1    = (const float*)d_in[10];
    const float*     ng1    = (const float*)d_in[11];
    const float*     nbeta1 = (const float*)d_in[12];
    const float*     nW2    = (const float*)d_in[13];
    const float*     nb2    = (const float*)d_in[14];

    float* nodeOut = (float*)d_out;
    float* edgeOut = nodeOut + (long long)N * 128;

    // workspace layout: agg (N*128 f32) | packed bf16 weights
    char* ws = (char*)d_ws;
    float* agg = (float*)ws;
    size_t off = (size_t)N * 128 * sizeof(float);
    uint32_t* peW1 = (uint32_t*)(ws + off); off += (size_t)384 * 256 * 2;
    uint32_t* peW2 = (uint32_t*)(ws + off); off += (size_t)256 * 128 * 2;
    uint32_t* pnW1 = (uint32_t*)(ws + off); off += (size_t)256 * 256 * 2;
    uint32_t* pnW2 = (uint32_t*)(ws + off); off += (size_t)256 * 128 * 2;

    long long aggN = (long long)N * 128;
    zero_f32_kernel<<<(int)((aggN + 255) / 256), 256, 0, stream>>>(agg, aggN);
    pack_weights_kernel<<<96, 256, 0, stream>>>(eW1, peW1, 384, 256);
    pack_weights_kernel<<<32, 256, 0, stream>>>(eW2, peW2, 256, 128);
    pack_weights_kernel<<<64, 256, 0, stream>>>(nW1, pnW1, 256, 256);
    pack_weights_kernel<<<32, 256, 0, stream>>>(nW2, pnW2, 256, 128);

    edge_mlp_kernel<<<(E + 31) / 32, 256, 0, stream>>>(
        nodeF, edgeF, eidx, eidx + E, peW1, eb1, eg1, ebeta1, peW2, eb2,
        edgeOut, agg, E);

    node_mlp_kernel<<<(N + 31) / 32, 256, 0, stream>>>(
        nodeF, agg, pnW1, nb1, ng1, nbeta1, pnW2, nb2, nodeOut, N);
}
